// CVMI_20426864460180
// MI455X (gfx1250) — matrix operand order repaired and checked by reading, not compiled, not run on hardware
//
#include <hip/hip_runtime.h>
#include <hip/hip_bf16.h>
#include <stdint.h>

// ---------------------------------------------------------------------------
// Cross width-axis attention (left/right) for MI455X (gfx1250, wave32, WMMA).
// Pipeline: f32->bf16 prep  ->  conv3x3 via implicit-GEMM bf16 WMMA
//           -> per-(b,h) score GEMM + dual softmax + dual P*V GEMM (bf16 WMMA)
// Conv LDS staging uses GLOBAL_LOAD_ASYNC_TO_LDS_B128 (ASYNCcnt path).
// ---------------------------------------------------------------------------

typedef __bf16 bf16_t;
typedef __attribute__((ext_vector_type(16))) __bf16 v16bf;
typedef __attribute__((ext_vector_type(8)))  __bf16 v8bf;
typedef __attribute__((ext_vector_type(8)))  float   v8f;
typedef __attribute__((ext_vector_type(4)))  float   v4f;

#define B_    4
#define C_    96
#define H_    192
#define W_    192
#define NELEM (B_*C_*H_*W_)   // 14,155,776 elements per tensor
#define KCONV (C_*9)          // 864 = implicit-GEMM K for 3x3 conv
#define SP    196             // padded score-row stride in floats (bank spread)

// Build a 16-element bf16 A/B fragment from two contiguous 8-element chunks.
static __device__ __forceinline__ v16bf ld2x8(const bf16_t* p0, const bf16_t* p1) {
  v8bf a = *(const v8bf*)p0;
  v8bf b = *(const v8bf*)p1;
  v16bf r;
#pragma unroll
  for (int i = 0; i < 8; ++i) { r[i] = a[i]; r[i + 8] = b[i]; }
  return r;
}

static __device__ __forceinline__ v8f wmma_bf16(v16bf a, v16bf b, v8f c) {
  return __builtin_amdgcn_wmma_f32_16x16x32_bf16(false, a, false, b,
                                                 (short)0, c, false, false);
}

// Async global->LDS 16B transfer (per active lane), tracked by ASYNCcnt.
static __device__ __forceinline__ void async_load_b128(unsigned lds_off,
                                                       const void* g) {
  asm volatile("global_load_async_to_lds_b128 %0, %1, off"
               :
               : "v"(lds_off), "v"((unsigned long long)(uintptr_t)g)
               : "memory");
}
static __device__ __forceinline__ void wait_async0() {
  asm volatile("s_wait_asynccnt 0x0" ::: "memory");
}

// ---------------------------------------------------------------------------
// Prep: f32 NCHW -> bf16 in BOTH layouts.
//   *_cl : channels-last [b][h][w][c]  (conv input / Q,K score operands)
//   *_r  : NCHW [b][c][h][w]           (V operand of the P*V GEMMs)
// Strided cl stores are absorbed by the 192 MB L2 (whole pass working set fits).
// ---------------------------------------------------------------------------
__global__ void prep_inputs(const float* __restrict__ low1,
                            const float* __restrict__ low2,
                            bf16_t* __restrict__ low1cl, bf16_t* __restrict__ low2cl,
                            bf16_t* __restrict__ low1r,  bf16_t* __restrict__ low2r) {
  for (int idx = blockIdx.x * blockDim.x + threadIdx.x; idx < 2 * NELEM;
       idx += gridDim.x * blockDim.x) {
    int t = idx / NELEM;
    int r = idx % NELEM;
    int w = r % W_;
    int h = (r / W_) % H_;
    int c = (r / (W_ * H_)) % C_;
    int b = r / (W_ * H_ * C_);
    float  v  = t ? low2[r] : low1[r];
    bf16_t bv = (bf16_t)v;
    (t ? low2r  : low1r )[r] = bv;
    (t ? low2cl : low1cl)[((b * H_ + h) * W_ + w) * C_ + c] = bv;
  }
}

// conv_w [O,I,3,3] -> wA [O][ (kh*3+kw)*96 + i ]  (A-operand row-major, bf16)
__global__ void prep_weights(const float* __restrict__ cw, bf16_t* __restrict__ wA) {
  int idx = blockIdx.x * blockDim.x + threadIdx.x;
  if (idx >= C_ * KCONV) return;
  int o   = idx / KCONV;
  int rem = idx % KCONV;
  int ci  = rem / 9;
  int kh  = (rem % 9) / 3;
  int kw  = rem % 3;
  wA[o * KCONV + (kh * 3 + kw) * C_ + ci] =
      (bf16_t)cw[((o * C_ + ci) * 3 + kh) * 3 + kw];
}

// ---------------------------------------------------------------------------
// Conv3x3 SAME as implicit GEMM: M=96 oc (6 tiles), N=16 pixels, K=864.
// One block per (16-wide pixel strip, h, b, tensor); 6 waves = 6 oc tiles.
// Input patch (3 rows x 18 wpos x 96 ch, zero-padded halo) staged in LDS via
// async-to-LDS loads; halo lanes zero-fill disjoint LDS bytes via DS stores.
// ---------------------------------------------------------------------------
__global__ __launch_bounds__(192) void conv3x3_wmma(
    const bf16_t* __restrict__ low1cl, const bf16_t* __restrict__ low2cl,
    const bf16_t* __restrict__ wA, const float* __restrict__ conv_b,
    bf16_t* __restrict__ Qcl, bf16_t* __restrict__ Kcl) {
  __shared__ bf16_t patch[3 * 18 * C_];   // 10,368 B

  const int w0 = blockIdx.x * 16;
  const int h  = blockIdx.y;
  const int b  = blockIdx.z >> 1;
  const int t  = blockIdx.z & 1;
  const bf16_t* in  = t ? low2cl : low1cl;
  bf16_t*       out = t ? Kcl    : Qcl;

  const int tid = threadIdx.x;
  // LDS-relative base of patch: flat LDS-aperture addresses carry the
  // workgroup-relative offset in addr[31:0] (ISA 10.2).
  const unsigned ldsb = (unsigned)(uintptr_t)(void*)patch;

  // Cooperative zero-padded halo fill: 648 chunks of 8 bf16 (16 B each).
  for (int ch = tid; ch < 3 * 18 * 12; ch += 192) {
    int c0 = (ch % 12) * 8;
    int wp = (ch / 12) % 18;
    int r  = ch / (12 * 18);
    int hh = h + r - 1;
    int ww = w0 + wp - 1;
    const int pidx = (r * 18 + wp) * C_ + c0;
    if (hh >= 0 && hh < H_ && ww >= 0 && ww < W_) {
      async_load_b128(ldsb + (unsigned)(pidx * 2),
                      in + ((b * H_ + hh) * W_ + ww) * C_ + c0);
    } else {
      v8bf z;
#pragma unroll
      for (int i = 0; i < 8; ++i) z[i] = (bf16_t)0.f;
      *(v8bf*)(patch + pidx) = z;               // disjoint bytes vs async writes
    }
  }
  wait_async0();        // ASYNCcnt==0: our async transfers have landed in LDS
  __syncthreads();

  const int wave = tid >> 5;
  const int lane = tid & 31;
  const int l16  = lane & 15;
  const int g    = (lane >> 4) & 1;
  const int m    = wave;                 // oc tile 0..5

  const bf16_t* aRow = wA + (m * 16 + l16) * KCONV;  // A lane row = oc
  v8f acc = {0.f, 0.f, 0.f, 0.f, 0.f, 0.f, 0.f, 0.f};

#pragma unroll
  for (int kh = 0; kh < 3; ++kh)
#pragma unroll
    for (int kw = 0; kw < 3; ++kw)
#pragma unroll
      for (int cc = 0; cc < 3; ++cc) {
        const int k0 = (kh * 3 + kw) * C_ + cc * 32;
        v16bf a = ld2x8(aRow + k0 + g * 8, aRow + k0 + 16 + g * 8);
        const bf16_t* bp = patch + (kh * 18 + (l16 + kw)) * C_ + cc * 32 + g * 8;
        v16bf bb = ld2x8(bp, bp + 16);
        acc = wmma_bf16(a, bb, acc);
      }

  // D tile: lane = pixel (N), 8 consecutive oc (M) per lane -> channels-last.
  const int ocD = m * 16 + g * 8;
  v8bf res;
#pragma unroll
  for (int i = 0; i < 8; ++i) res[i] = (bf16_t)(acc[i] + conv_b[ocD + i]);
  *(v8bf*)(out + ((b * H_ + h) * W_ + (w0 + l16)) * C_ + ocD) = res;
}

// ---------------------------------------------------------------------------
// Attention per (b,h): S = Q^T K (192x192, K=96) -> dual softmax -> P*V x2.
// 256 threads = 8 waves; full f32 score tile lives in LDS (~150 KB / 320 KB,
// 2 workgroups per WGP).
// ---------------------------------------------------------------------------
__global__ __launch_bounds__(256) void attn_wmma(
    const bf16_t* __restrict__ Qcl, const bf16_t* __restrict__ Kcl,
    const bf16_t* __restrict__ low1r, const bf16_t* __restrict__ low2r,
    const float* __restrict__ low1, const float* __restrict__ low2,
    float* __restrict__ outL, float* __restrict__ outR) {
  extern __shared__ float smem[];
  float* Sm     = smem;                  // [192][SP]
  float* rowmax = smem + W_ * SP;
  float* rowrcp = rowmax + W_;
  float* colmax = rowrcp + W_;
  float* colrcp = colmax + W_;

  const int h = blockIdx.x;
  const int b = blockIdx.y;
  const bf16_t* Qr = Qcl + (b * H_ + h) * W_ * C_;   // [w][c]
  const bf16_t* Kr = Kcl + (b * H_ + h) * W_ * C_;   // [v][c]

  const int tid  = threadIdx.x;
  const int wave = tid >> 5;
  const int lane = tid & 31;
  const int l16  = lane & 15;
  const int g    = (lane >> 4) & 1;

  // ---- Phase A: score tiles (12x12 tiles of 16x16, 3 bf16 WMMA each) ----
  for (int t = wave; t < 144; t += 8) {
    const int ti = t / 12, tj = t % 12;
    const bf16_t* aRow = Qr + (ti * 16 + l16) * C_;  // A lane row = w
    const bf16_t* bCol = Kr + (tj * 16 + l16) * C_;  // B lane col = v
    v8f acc = {0.f, 0.f, 0.f, 0.f, 0.f, 0.f, 0.f, 0.f};
#pragma unroll
    for (int cc = 0; cc < 3; ++cc) {
      const int c0 = cc * 32;
      v16bf a  = ld2x8(aRow + c0 + g * 8, aRow + c0 + 16 + g * 8);
      v16bf bb = ld2x8(bCol + c0 + g * 8, bCol + c0 + 16 + g * 8);
      acc = wmma_bf16(a, bb, acc);
    }
#pragma unroll
    for (int i = 0; i < 8; ++i)
      Sm[(ti * 16 + g * 8 + i) * SP + tj * 16 + l16] = acc[i];
  }
  __syncthreads();

  // ---- Phase B: softmax stats along both axes (relu(softmax)=softmax) ----
  if (tid < W_) {
    const int r = tid;
    const float* row = Sm + r * SP;       // 16B-aligned (SP*4 % 16 == 0)
    float m = -3.0e38f;
    for (int v = 0; v < W_; v += 8) {     // vectorized contiguous row scan
      v4f x0 = *(const v4f*)(row + v);
      v4f x1 = *(const v4f*)(row + v + 4);
      m = fmaxf(m, fmaxf(fmaxf(x0[0], x0[1]), fmaxf(x0[2], x0[3])));
      m = fmaxf(m, fmaxf(fmaxf(x1[0], x1[1]), fmaxf(x1[2], x1[3])));
    }
    float s = 0.f;
    for (int v = 0; v < W_; v += 8) {
      v4f x0 = *(const v4f*)(row + v);
      v4f x1 = *(const v4f*)(row + v + 4);
      s += __expf(x0[0] - m) + __expf(x0[1] - m) +
           __expf(x0[2] - m) + __expf(x0[3] - m) +
           __expf(x1[0] - m) + __expf(x1[1] - m) +
           __expf(x1[2] - m) + __expf(x1[3] - m);
    }
    rowmax[r] = m; rowrcp[r] = 1.f / s;

    float cm = -3.0e38f;
    for (int w = 0; w < W_; ++w) cm = fmaxf(cm, Sm[w * SP + r]);
    float cs = 0.f;
    for (int w = 0; w < W_; ++w) cs += __expf(Sm[w * SP + r] - cm);
    colmax[r] = cm; colrcp[r] = 1.f / cs;
  }
  __syncthreads();

  // ---- Phase C1: left = low1 + A_left * low2   (M=w 12 tiles, N=c 6, K=v 192)
  for (int m = wave; m < 12; m += 8) {
    const int wM = m * 16 + l16;                  // A lane row = w
    const float rm = rowmax[wM], rr = rowrcp[wM];
    v8f acc[6];
#pragma unroll
    for (int nt = 0; nt < 6; ++nt) acc[nt] = (v8f){0.f,0.f,0.f,0.f,0.f,0.f,0.f,0.f};
    for (int ks = 0; ks < 6; ++ks) {
      const int k0 = ks * 32;
      v16bf a;
#pragma unroll
      for (int j = 0; j < 8; ++j) {               // normalized exp -> bf16 A
        a[j]     = (bf16_t)(__expf(Sm[wM * SP + k0 +      g * 8 + j] - rm) * rr);
        a[j + 8] = (bf16_t)(__expf(Sm[wM * SP + k0 + 16 + g * 8 + j] - rm) * rr);
      }
#pragma unroll
      for (int nt = 0; nt < 6; ++nt) {
        const int c = nt * 16 + l16;              // B lane col = channel
        const bf16_t* bp = low2r + ((b * C_ + c) * H_ + h) * W_ + k0 + g * 8;
        __builtin_prefetch(bp + 32, 0, 1);
        acc[nt] = wmma_bf16(a, ld2x8(bp, bp + 16), acc[nt]);
      }
    }
#pragma unroll
    for (int nt = 0; nt < 6; ++nt) {              // residual add, 32B f32 stores
      const int c = nt * 16 + l16;
      const int base = ((b * C_ + c) * H_ + h) * W_ + m * 16 + g * 8;
      v8f prev = *(const v8f*)(low1 + base);
      *(v8f*)(outL + base) = prev + acc[nt];
    }
  }

  // ---- Phase C2: right = low2 + A_right * low1 (A_right = colsoftmax(S^T)) --
  for (int m = wave; m < 12; m += 8) {
    const int wM = m * 16 + l16;
    const float cm = colmax[wM], cr = colrcp[wM];
    v8f acc[6];
#pragma unroll
    for (int nt = 0; nt < 6; ++nt) acc[nt] = (v8f){0.f,0.f,0.f,0.f,0.f,0.f,0.f,0.f};
    for (int ks = 0; ks < 6; ++ks) {
      const int k0 = ks * 32;
      v16bf a;
#pragma unroll
      for (int j = 0; j < 8; ++j) {               // transposed read of S
        a[j]     = (bf16_t)(__expf(Sm[(k0 +      g * 8 + j) * SP + wM] - cm) * cr);
        a[j + 8] = (bf16_t)(__expf(Sm[(k0 + 16 + g * 8 + j) * SP + wM] - cm) * cr);
      }
#pragma unroll
      for (int nt = 0; nt < 6; ++nt) {
        const int c = nt * 16 + l16;
        const bf16_t* bp = low1r + ((b * C_ + c) * H_ + h) * W_ + k0 + g * 8;
        acc[nt] = wmma_bf16(a, ld2x8(bp, bp + 16), acc[nt]);
      }
    }
#pragma unroll
    for (int nt = 0; nt < 6; ++nt) {
      const int c = nt * 16 + l16;
      const int base = ((b * C_ + c) * H_ + h) * W_ + m * 16 + g * 8;
      v8f prev = *(const v8f*)(low2 + base);
      *(v8f*)(outR + base) = prev + acc[nt];
    }
  }
}

// ---------------------------------------------------------------------------
extern "C" void kernel_launch(void* const* d_in, const int* in_sizes, int n_in,
                              void* d_out, int out_size, void* d_ws, size_t ws_size,
                              hipStream_t stream) {
  (void)in_sizes; (void)n_in; (void)out_size; (void)ws_size;
  const float* low1   = (const float*)d_in[0];
  const float* low2   = (const float*)d_in[1];
  const float* conv_w = (const float*)d_in[2];
  const float* conv_b = (const float*)d_in[3];
  float* out = (float*)d_out;

  char*  ws = (char*)d_ws;
  const size_t nb = (size_t)NELEM * sizeof(bf16_t);  // one bf16 tensor
  bf16_t* low1cl = (bf16_t*)(ws);
  bf16_t* low2cl = (bf16_t*)(ws + 1 * nb);
  bf16_t* low1r  = (bf16_t*)(ws + 2 * nb);
  bf16_t* low2r  = (bf16_t*)(ws + 3 * nb);
  bf16_t* Qcl    = (bf16_t*)(ws + 4 * nb);
  bf16_t* Kcl    = (bf16_t*)(ws + 5 * nb);
  bf16_t* wA     = (bf16_t*)(ws + 6 * nb);           // 162 KB, L2-resident

  prep_inputs<<<2048, 256, 0, stream>>>(low1, low2, low1cl, low2cl, low1r, low2r);
  prep_weights<<<(C_ * KCONV + 255) / 256, 256, 0, stream>>>(conv_w, wA);

  conv3x3_wmma<<<dim3(W_ / 16, H_, B_ * 2), 192, 0, stream>>>(
      low1cl, low2cl, wA, conv_b, Qcl, Kcl);

  const size_t smem = (size_t)(W_ * SP + 4 * W_) * sizeof(float);  // 153,600 B
  hipFuncSetAttribute((const void*)attn_wmma,
                      hipFuncAttributeMaxDynamicSharedMemorySize, (int)smem);
  attn_wmma<<<dim3(H_, B_), 256, smem, stream>>>(
      Qcl, Kcl, low1r, low2r, low1, low2, out, out + NELEM);
}
